// MutPredV2Model_87771951661799
// MI455X (gfx1250) — matrix operand order repaired and checked by reading, not compiled, not run on hardware
//
#include <hip/hip_runtime.h>
#include <stdint.h>

typedef __attribute__((ext_vector_type(16))) __bf16 v16bf;
typedef __attribute__((ext_vector_type(2)))  __bf16 v2bf;
typedef __attribute__((ext_vector_type(8)))  float  v8f;

#define NND  20000
#define NED  320000
#define SDIM 128
#define EDIM 1280
#define EDGD 64
#define HD   256
#define NL   4
#define NC   20

#define LDA 40    // padded A stride in ushorts (80B/row, 16B aligned)
#define LDB 264   // padded B stride in ushorts (528B/row, 16B aligned)

// Native RNE conversions: let the backend pick v_cvt_*bf16* forms.
__device__ __forceinline__ unsigned short f2bf(float x) {
  union { __bf16 b; unsigned short u; } cv;
  cv.b = (__bf16)x;
  return cv.u;
}

__device__ __forceinline__ unsigned pack2bf(float lo, float hi) {
  union { v2bf v; unsigned u; } cv;
  cv.v = (v2bf){(__bf16)lo, (__bf16)hi};
  return cv.u;
}

// Convert 8 consecutive floats (16B-aligned) -> 8 bf16, one b128 LDS store.
__device__ __forceinline__ void cvt8_store(unsigned short* dst, const float* src) {
  float4 a = ((const float4*)src)[0];
  float4 b = ((const float4*)src)[1];
  uint4 o;
  o.x = pack2bf(a.x, a.y);
  o.y = pack2bf(a.z, a.w);
  o.z = pack2bf(b.x, b.y);
  o.w = pack2bf(b.z, b.w);
  *(uint4*)dst = o;
}

union FragU { uint4 u[2]; v16bf v; };

// A fragment (16x32 bf16): lanes 0-15 row M=l, K chunks [0..7]+[16..23];
// lanes 16-31 row M=l-16, K chunks [8..15]+[24..31].
__device__ __forceinline__ v16bf load_afrag(const unsigned short* As, int rowBase, int lane) {
  int r  = rowBase + (lane & 15);
  int kb = (lane >> 4) * 8;
  FragU f;
  f.u[0] = *(const uint4*)(As + r * LDA + kb);
  f.u[1] = *(const uint4*)(As + r * LDA + kb + 16);
  return f.v;
}

// B fragment (32x16 bf16): lanes 0-15 col N=l, K [0..7]+[16..23];
// lanes 16-31 col N=l-16, K [8..15]+[24..31].
__device__ __forceinline__ v16bf load_bfrag(const unsigned short* Bs, int col, int lane) {
  union { unsigned short s[16]; v16bf v; } f;
  const unsigned short* p = Bs + ((lane >> 4) * 8) * LDB + col + (lane & 15);
#pragma unroll
  for (int i = 0; i < 8; i++) {
    f.s[i]     = p[i * LDB];
    f.s[8 + i] = p[(16 + i) * LDB];
  }
  return f.v;
}

// Stage 32xHD bf16 weight slice into LDS (256 threads).
__device__ __forceinline__ void stageB(unsigned short* Bs, const unsigned short* Wg) {
  int t = threadIdx.x;
  int r = t >> 3, c0 = (t & 7) * 32;
  const uint4* s = (const uint4*)(Wg + r * HD + c0);
  uint4* d = (uint4*)(Bs + r * LDB + c0);
  d[0] = s[0]; d[1] = s[1]; d[2] = s[2]; d[3] = s[3];
}

__device__ __forceinline__ void mma_step(const unsigned short* As, const unsigned short* Bs,
                                         int rowBase, int colBase, int lane, v8f acc[8]) {
  v16bf a = load_afrag(As, rowBase, lane);
#pragma unroll
  for (int nt = 0; nt < 8; nt++) {
    v16bf b = load_bfrag(Bs, colBase + nt * 16, lane);
    acc[nt] = __builtin_amdgcn_wmma_f32_16x16x32_bf16(false, a, false, b,
                                                      (short)0, acc[nt], false, false);
  }
}

__device__ __forceinline__ void zero_acc(v8f acc[8]) {
  v8f z = {0.f, 0.f, 0.f, 0.f, 0.f, 0.f, 0.f, 0.f};
#pragma unroll
  for (int i = 0; i < 8; i++) acc[i] = z;
}

// ---------------- embed: h = relu(xs@Ws + xe@We + bs + be) ----------------
__global__ __launch_bounds__(256) void embed_gemm(
    const float* __restrict__ xs, const float* __restrict__ xe,
    const unsigned short* __restrict__ Wemb,   // (SDIM+EDIM) x HD bf16
    const float* __restrict__ bs, const float* __restrict__ be,
    float* __restrict__ hout) {
  __shared__ unsigned short As[2][64 * LDA];
  __shared__ unsigned short Bs[2][32 * LDB];
  int m0 = blockIdx.x * 64;
  int t = threadIdx.x, lane = t & 31, w = t >> 5;
  int rowBase = (w & 3) * 16, colBase = (w >> 2) * 128;
  v8f acc[8]; zero_acc(acc);
  int r = t >> 2, c0 = (t & 3) * 8;
  int row = m0 + r; if (row >= NND) row = NND - 1;     // clamp; masked at store
  const int K = SDIM + EDIM;

  auto stage = [&](int ks, int bf) {
    int gk = ks + c0;                                  // chunk never straddles SDIM
    const float* src = (gk < SDIM) ? xs + (size_t)row * SDIM + gk
                                   : xe + (size_t)row * EDIM + (gk - SDIM);
    cvt8_store(&As[bf][r * LDA + c0], src);
    stageB(Bs[bf], Wemb + (size_t)ks * HD);
  };

  stage(0, 0);
  int cur = 0;
  for (int ks = 0; ks < K; ks += 32) {
    __syncthreads();
    if (ks + 32 < K) stage(ks + 32, cur ^ 1);
    mma_step(As[cur], Bs[cur], rowBase, colBase, lane, acc);
    cur ^= 1;
  }
#pragma unroll
  for (int rr = 0; rr < 8; rr++) {
    int orow = m0 + rowBase + (lane >> 4) * 8 + rr;
    if (orow < NND) {
#pragma unroll
      for (int nt = 0; nt < 8; nt++) {
        int col = colBase + nt * 16 + (lane & 15);
        float v = acc[nt][rr] + bs[col] + be[col];
        hout[(size_t)orow * HD + col] = v > 0.f ? v : 0.f;
      }
    }
  }
}

// ------- msg1: m1 = bf16(relu(concat(h[src], eattr) @ W1 + b1)) -----------
__global__ __launch_bounds__(256) void msg1_gemm(
    const float* __restrict__ h, const float* __restrict__ eattr,
    const int* __restrict__ srcIdx,
    const unsigned short* __restrict__ W1, const float* __restrict__ b1,
    unsigned short* __restrict__ m1) {
  __shared__ unsigned short As[2][64 * LDA];
  __shared__ unsigned short Bs[2][32 * LDB];
  int m0 = blockIdx.x * 64;
  int t = threadIdx.x, lane = t & 31, w = t >> 5;
  int rowBase = (w & 3) * 16, colBase = (w >> 2) * 128;
  v8f acc[8]; zero_acc(acc);
  int r = t >> 2, c0 = (t & 3) * 8;
  int e = m0 + r;                       // NED is a multiple of 64
  int s = srcIdx[e];
  const int K = HD + EDGD;

  auto stage = [&](int ks, int bf) {
    int gk = ks + c0;                                  // chunk never straddles HD
    const float* src = (gk < HD) ? h + (size_t)s * HD + gk
                                 : eattr + (size_t)e * EDGD + (gk - HD);
    cvt8_store(&As[bf][r * LDA + c0], src);
    stageB(Bs[bf], W1 + (size_t)ks * HD);
  };

  stage(0, 0);
  int cur = 0;
  for (int ks = 0; ks < K; ks += 32) {
    __syncthreads();
    if (ks + 32 < K) stage(ks + 32, cur ^ 1);
    mma_step(As[cur], Bs[cur], rowBase, colBase, lane, acc);
    cur ^= 1;
  }
#pragma unroll
  for (int rr = 0; rr < 8; rr++) {
    int orow = m0 + rowBase + (lane >> 4) * 8 + rr;
#pragma unroll
    for (int nt = 0; nt < 8; nt++) {
      int col = colBase + nt * 16 + (lane & 15);
      float v = acc[nt][rr] + b1[col];
      m1[(size_t)orow * HD + col] = f2bf(v > 0.f ? v : 0.f);
    }
  }
}

// --- msg2: agg[dst[e]] += relu(m1 @ W2 + b2)  (fused segment_sum) ---------
__global__ __launch_bounds__(256) void msg2_gemm(
    const unsigned short* __restrict__ m1, const int* __restrict__ dstIdx,
    const unsigned short* __restrict__ W2, const float* __restrict__ b2,
    float* __restrict__ agg) {
  __shared__ unsigned short As[2][64 * LDA];
  __shared__ unsigned short Bs[2][32 * LDB];
  int m0 = blockIdx.x * 64;
  int t = threadIdx.x, lane = t & 31, w = t >> 5;
  int rowBase = (w & 3) * 16, colBase = (w >> 2) * 128;
  v8f acc[8]; zero_acc(acc);
  int r = t >> 2, c0 = (t & 3) * 8;
  int e = m0 + r;

  auto stage = [&](int ks, int bf) {
    *(uint4*)(&As[bf][r * LDA + c0]) = *(const uint4*)(m1 + (size_t)e * HD + ks + c0);
    stageB(Bs[bf], W2 + (size_t)ks * HD);
  };

  stage(0, 0);
  int cur = 0;
  for (int ks = 0; ks < HD; ks += 32) {
    __syncthreads();
    if (ks + 32 < HD) stage(ks + 32, cur ^ 1);
    mma_step(As[cur], Bs[cur], rowBase, colBase, lane, acc);
    cur ^= 1;
  }
#pragma unroll
  for (int rr = 0; rr < 8; rr++) {
    int orow = m0 + rowBase + (lane >> 4) * 8 + rr;
    int d = dstIdx[orow];
    float* ap = agg + (size_t)d * HD + colBase + (lane & 15);
#pragma unroll
    for (int nt = 0; nt < 8; nt++) {
      int col = colBase + nt * 16 + (lane & 15);
      float v = acc[nt][rr] + b2[col];
      unsafeAtomicAdd(ap + nt * 16, v > 0.f ? v : 0.f);
    }
  }
}

// ------- upd: hnew = relu(concat(h, agg) @ Wu + bu) -----------------------
__global__ __launch_bounds__(256) void upd_gemm(
    const float* __restrict__ h, const float* __restrict__ agg,
    const unsigned short* __restrict__ Wu, const float* __restrict__ bu,
    float* __restrict__ hnew) {
  __shared__ unsigned short As[2][64 * LDA];
  __shared__ unsigned short Bs[2][32 * LDB];
  int m0 = blockIdx.x * 64;
  int t = threadIdx.x, lane = t & 31, w = t >> 5;
  int rowBase = (w & 3) * 16, colBase = (w >> 2) * 128;
  v8f acc[8]; zero_acc(acc);
  int r = t >> 2, c0 = (t & 3) * 8;
  int row = m0 + r; if (row >= NND) row = NND - 1;     // clamp; masked at store
  const int K = 2 * HD;

  auto stage = [&](int ks, int bf) {
    int gk = ks + c0;                                  // chunk never straddles HD
    const float* src = (gk < HD) ? h + (size_t)row * HD + gk
                                 : agg + (size_t)row * HD + (gk - HD);
    cvt8_store(&As[bf][r * LDA + c0], src);
    stageB(Bs[bf], Wu + (size_t)ks * HD);
  };

  stage(0, 0);
  int cur = 0;
  for (int ks = 0; ks < K; ks += 32) {
    __syncthreads();
    if (ks + 32 < K) stage(ks + 32, cur ^ 1);
    mma_step(As[cur], Bs[cur], rowBase, colBase, lane, acc);
    cur ^= 1;
  }
#pragma unroll
  for (int rr = 0; rr < 8; rr++) {
    int orow = m0 + rowBase + (lane >> 4) * 8 + rr;
    if (orow < NND) {
#pragma unroll
      for (int nt = 0; nt < 8; nt++) {
        int col = colBase + nt * 16 + (lane & 15);
        float v = acc[nt][rr] + bu[col];
        hnew[(size_t)orow * HD + col] = v > 0.f ? v : 0.f;
      }
    }
  }
}

// ------- LayerNorm over HD=256, one wave32 per row; optional residual -----
__global__ __launch_bounds__(256) void ln_kernel(
    const float* __restrict__ xin, const float* __restrict__ res,
    const float* __restrict__ g, const float* __restrict__ b,
    float* __restrict__ out, int n) {
  int row = blockIdx.x * 8 + (threadIdx.x >> 5);
  int lane = threadIdx.x & 31;
  if (row >= n) return;
  float v[8];
  float s = 0.f;
#pragma unroll
  for (int i = 0; i < 8; i++) {
    int c = lane + 32 * i;
    float x = xin[(size_t)row * HD + c];
    if (res) x += res[(size_t)row * HD + c];
    v[i] = x; s += x;
  }
#pragma unroll
  for (int off = 16; off > 0; off >>= 1) s += __shfl_xor(s, off, 32);
  float mu = s * (1.f / HD);
  float var = 0.f;
#pragma unroll
  for (int i = 0; i < 8; i++) { float d = v[i] - mu; var += d * d; }
#pragma unroll
  for (int off = 16; off > 0; off >>= 1) var += __shfl_xor(var, off, 32);
  float rs = rsqrtf(var * (1.f / HD) + 1e-5f);
#pragma unroll
  for (int i = 0; i < 8; i++) {
    int c = lane + 32 * i;
    out[(size_t)row * HD + c] = (v[i] - mu) * rs * g[c] + b[c];
  }
}

// ------- head: out = h @ head_W + head_b  (tiny, scalar) ------------------
__global__ __launch_bounds__(256) void head_kernel(
    const float* __restrict__ h, const float* __restrict__ W,
    const float* __restrict__ b, float* __restrict__ out, int n) {
  int i = blockIdx.x * 256 + threadIdx.x;
  if (i >= n * NC) return;
  int row = i / NC, c = i % NC;
  float s = b[c];
  const float* hr = h + (size_t)row * HD;
#pragma unroll 8
  for (int k = 0; k < HD; k++) s += hr[k] * W[k * NC + c];
  out[i] = s;
}

__global__ __launch_bounds__(256) void cvt_bf16(
    const float* __restrict__ s, unsigned short* __restrict__ d, int n) {
  int i = blockIdx.x * 256 + threadIdx.x;
  if (i < n) d[i] = f2bf(s[i]);
}

extern "C" void kernel_launch(void* const* d_in, const int* in_sizes, int n_in,
                              void* d_out, int out_size, void* d_ws, size_t ws_size,
                              hipStream_t stream) {
  const float* x_struct = (const float*)d_in[0];
  const float* x_esm    = (const float*)d_in[1];
  const int*   eidx     = (const int*)d_in[2];
  const float* eattr    = (const float*)d_in[3];
  const float* W_struct = (const float*)d_in[4];
  const float* b_struct = (const float*)d_in[5];
  const float* W_esm    = (const float*)d_in[6];
  const float* b_esm    = (const float*)d_in[7];
  const float* msg_W1   = (const float*)d_in[8];
  const float* msg_b1   = (const float*)d_in[9];
  const float* msg_W2   = (const float*)d_in[10];
  const float* msg_b2   = (const float*)d_in[11];
  const float* upd_W    = (const float*)d_in[12];
  const float* upd_b    = (const float*)d_in[13];
  const float* ln_g     = (const float*)d_in[14];
  const float* ln_b     = (const float*)d_in[15];
  const float* fn_g     = (const float*)d_in[16];
  const float* fn_b     = (const float*)d_in[17];
  const float* head_W   = (const float*)d_in[18];
  const float* head_b   = (const float*)d_in[19];

  const int* srcI = eidx;
  const int* dstI = eidx + NED;

  char* p = (char*)d_ws;
  float* h    = (float*)p; p += (size_t)NND * HD * 4;
  float* hnew = (float*)p; p += (size_t)NND * HD * 4;
  float* agg  = (float*)p; p += (size_t)NND * HD * 4;
  unsigned short* m1   = (unsigned short*)p; p += (size_t)NED * HD * 2;
  unsigned short* Wemb = (unsigned short*)p; p += (size_t)(SDIM + EDIM) * HD * 2;
  unsigned short* Wm1  = (unsigned short*)p; p += (size_t)NL * (HD + EDGD) * HD * 2;
  unsigned short* Wm2  = (unsigned short*)p; p += (size_t)NL * HD * HD * 2;
  unsigned short* Wu   = (unsigned short*)p; p += (size_t)NL * 2 * HD * HD * 2;
  (void)ws_size; (void)in_sizes; (void)n_in; (void)out_size;

  // Pre-convert weights to bf16 (small, done once per launch).
  {
    int n;
    n = SDIM * HD;              cvt_bf16<<<(n + 255) / 256, 256, 0, stream>>>(W_struct, Wemb, n);
    n = EDIM * HD;              cvt_bf16<<<(n + 255) / 256, 256, 0, stream>>>(W_esm, Wemb + SDIM * HD, n);
    n = NL * (HD + EDGD) * HD;  cvt_bf16<<<(n + 255) / 256, 256, 0, stream>>>(msg_W1, Wm1, n);
    n = NL * HD * HD;           cvt_bf16<<<(n + 255) / 256, 256, 0, stream>>>(msg_W2, Wm2, n);
    n = NL * 2 * HD * HD;       cvt_bf16<<<(n + 255) / 256, 256, 0, stream>>>(upd_W, Wu, n);
  }

  embed_gemm<<<(NND + 63) / 64, 256, 0, stream>>>(x_struct, x_esm, Wemb, b_struct, b_esm, h);

  for (int l = 0; l < NL; l++) {
    msg1_gemm<<<NED / 64, 256, 0, stream>>>(h, eattr, srcI,
        Wm1 + (size_t)l * (HD + EDGD) * HD, msg_b1 + l * HD, m1);
    hipMemsetAsync(agg, 0, (size_t)NND * HD * 4, stream);
    msg2_gemm<<<NED / 64, 256, 0, stream>>>(m1, dstI,
        Wm2 + (size_t)l * HD * HD, msg_b2 + l * HD, agg);
    upd_gemm<<<(NND + 63) / 64, 256, 0, stream>>>(h, agg,
        Wu + (size_t)l * 2 * HD * HD, upd_b + l * HD, hnew);
    ln_kernel<<<(NND + 7) / 8, 256, 0, stream>>>(hnew, h, ln_g + l * HD, ln_b + l * HD, h, NND);
  }
  ln_kernel<<<(NND + 7) / 8, 256, 0, stream>>>(h, nullptr, fn_g, fn_b, h, NND);
  head_kernel<<<(NND * NC + 255) / 256, 256, 0, stream>>>(h, head_W, head_b, (float*)d_out, NND);
}
